// MTRNNCell_12970801233939
// MI455X (gfx1250) — matrix operand order repaired and checked by reading, not compiled, not run on hardware
//
#include <hip/hip_runtime.h>
#include <math.h>

// ---------------------------------------------------------------------------
// MTRNN cell for MI455X (gfx1250, wave32, WMMA).
// GEMMs run as bf16 x bf16 -> f32 via v_wmma_f32_16x16x32_bf16 (roofline-
// balanced on MI455X: ~140 FLOP/byte workload vs ~100 FLOP/byte bf16 ridge).
// Staging uses GLOBAL_LOAD_ASYNC_TO_LDS (ASYNCcnt) with double-buffered LDS
// when the toolchain exposes the builtin; falls back to synchronous staging.
// ---------------------------------------------------------------------------

typedef __attribute__((ext_vector_type(16))) __bf16 v16bf;
typedef __attribute__((ext_vector_type(8)))  float  v8f;
typedef int vint4 __attribute__((vector_size(16)));   // builtin's pointee type

#define BROWS      4096
#define OUT_STRIDE 9728
#define U_SECT     4864

#ifndef __has_builtin
#define __has_builtin(x) 0
#endif
#if __has_builtin(__builtin_amdgcn_global_load_async_to_lds_b128) && \
    __has_builtin(__builtin_amdgcn_s_wait_asynccnt)
#define USE_ASYNC_LDS 1
#else
#define USE_ASYNC_LDS 0
#endif

#define AS1 __attribute__((address_space(1)))
#define AS3 __attribute__((address_space(3)))

static __device__ __forceinline__ unsigned short f2bf_rne(float f) {
  unsigned int u = __builtin_bit_cast(unsigned int, f);
  u += 0x7FFFu + ((u >> 16) & 1u);
  return (unsigned short)(u >> 16);
}

// Branchless tanh: 1 - 2/(e^{2x}+1).  v_exp_f32 + v_rcp_f32, no saveexec
// divergence; |err| ~1e-6, far below the bf16 GEMM quantization (~1e-2).
static __device__ __forceinline__ float tanh_fast(float x) {
  float e = __expf(2.0f * x);
  float r = __builtin_amdgcn_rcpf(e + 1.0f);
  return fmaf(-2.0f, r, 1.0f);
}

struct Frag32 { uint4 a, b; };  // 32 bytes == one v16bf fragment

// ---------------------------- attention kernel -----------------------------
__global__ void att_kernel(const float* __restrict__ hsl,
                           const float* __restrict__ hsr,
                           const float* __restrict__ katt,
                           const float* __restrict__ batt,
                           float* __restrict__ attl,
                           float* __restrict__ attr)
{
  const float* hs  = blockIdx.y ? hsr  : hsl;
  float*       att = blockIdx.y ? attr : attl;
  int r = blockIdx.x * blockDim.x + threadIdx.x;
  if (r >= BROWS) return;
  float a0 = batt[0], a1 = batt[1], a2 = batt[2];
  const float* h = hs + (size_t)r * 256;
  #pragma unroll 4
  for (int k = 0; k < 256; ++k) {
    float hv = h[k];
    a0 = fmaf(hv, katt[3 * k + 0], a0);
    a1 = fmaf(hv, katt[3 * k + 1], a1);
    a2 = fmaf(hv, katt[3 * k + 2], a2);
  }
  a0 = 1.f / (1.f + __expf(-a0));
  a1 = 1.f / (1.f + __expf(-a1));
  a2 = 1.f / (1.f + __expf(-a2));
  float inv = 1.f / (a0 + a1 + a2 + 0.01f);
  att[3 * r + 0] = a0 * inv;
  att[3 * r + 1] = a1 * inv;
  att[3 * r + 2] = a2 * inv;
}

// ----------------------------- pack A (f32->bf16) --------------------------
struct PackAJob { const float* src; unsigned short* dst; const float* att; int attIdx; int cshift; };
struct PackAArgs { PackAJob j[25]; };

__global__ void packA_kernel(PackAArgs args)
{
  PackAJob jb = args.j[blockIdx.y];
  int total = BROWS << jb.cshift;
  int idx = blockIdx.x * 256 + threadIdx.x;
  if (idx >= total) return;
  float v = jb.src[idx];
  if (jb.att) v *= jb.att[(idx >> jb.cshift) * 3 + jb.attIdx];
  jb.dst[idx] = f2bf_rne(v);
}

// --------------------- pack W (f32 [K,N] -> bf16 Wt[N,K]) ------------------
struct PackWJob { const float* w; unsigned short* wt; int kshift; int N; };
struct PackWArgs { PackWJob j[39]; };

__global__ void packW_kernel(PackWArgs args)
{
  PackWJob jb = args.j[blockIdx.y];
  int total = jb.N << jb.kshift;
  int idx = blockIdx.x * 256 + threadIdx.x;
  if (idx >= total) return;
  int n = idx >> jb.kshift;
  int k = idx & ((1 << jb.kshift) - 1);
  jb.wt[idx] = f2bf_rne(jb.w[(size_t)k * jb.N + n]);
}

// --------------------------- segmented WMMA GEMM ---------------------------
// C[4096 x N] = sum_seg A_seg[4096 x K] @ Wt_seg^T ;  fused leaky-int + tanh.
struct GemmSeg  { const unsigned short* A; const unsigned short* W; int K; };
struct GemmArgs {
  GemmSeg seg[6];
  int nseg;
  int N;                       // 512 or 256, multiple of 128
  const float* uprev;          // [4096, N]
  float c1, c2;
  float* out;                  // [4096, 9728]
  int hcol;                    // h column offset; u goes to hcol + 4864
};

// Workgroup: 256 threads = 8 waves; tile 64M x 128N; wave = 32x32 (2x2 WMMA).
// LDS rows padded to 40 halfwords (20 dwords) -> b128 fragment reads from 16
// lanes start at banks 0,20,40,... covering all 64 banks conflict-free.
// Double-buffered LDS, next tile staged (async) while WMMA consumes current.
__global__ __launch_bounds__(256)
void gemm_kernel(GemmArgs g)
{
  __shared__ unsigned short As[2 * 64 * 40];    // 10.0 KB
  __shared__ unsigned short Bs[2 * 128 * 40];   // 20.0 KB (of 320 KB/WGP)

  const int tid    = threadIdx.x;
  const int lane   = tid & 31;
  const int laneLo = lane & 15;
  const int half   = lane >> 4;
  const int wid    = tid >> 5;
  const int wm     = wid & 1;      // 2 waves along M
  const int wn     = wid >> 1;     // 4 waves along N
  const int blockM = blockIdx.x * 64;
  const int blockN = blockIdx.y * 128;

  v8f acc[2][2];
  #pragma unroll
  for (int i = 0; i < 2; ++i)
    #pragma unroll
    for (int j = 0; j < 2; ++j)
      #pragma unroll
      for (int v = 0; v < 8; ++v) acc[i][j][v] = 0.0f;

  const int arow = tid >> 2, acol = (tid & 3) * 8;   // A stage: 64 rows x 32 k
  const int brow = tid >> 1, bcol = (tid & 1) * 16;  // B stage: 128 rows x 32 k

  // Stage one 32-wide K tile of segment si at offset ki into buffer b.
  auto stage = [&](int si, int ki, int b) {
    const GemmSeg sg = g.seg[si];
    const unsigned short* ap = sg.A + (size_t)(blockM + arow) * sg.K + ki + acol;
    const unsigned short* wp = sg.W + (size_t)(blockN + brow) * sg.K + ki + bcol;
    unsigned short* la = &As[b * (64 * 40) + arow * 40 + acol];
    unsigned short* lb = &Bs[b * (128 * 40) + brow * 40 + bcol];
#if USE_ASYNC_LDS
    __builtin_amdgcn_global_load_async_to_lds_b128(
        (AS1 vint4*)(unsigned short*)ap,       (AS3 vint4*)la,       0, 0);
    __builtin_amdgcn_global_load_async_to_lds_b128(
        (AS1 vint4*)(unsigned short*)wp,       (AS3 vint4*)lb,       0, 0);
    __builtin_amdgcn_global_load_async_to_lds_b128(
        (AS1 vint4*)(unsigned short*)(wp + 8), (AS3 vint4*)(lb + 8), 0, 0);
#else
    *(uint4*)la = *(const uint4*)ap;
    ((uint4*)lb)[0]   = *(const uint4*)wp;
    *(uint4*)(lb + 8) = *(const uint4*)(wp + 8);
#endif
  };

  int total = 0;
  for (int t = 0; t < g.nseg; ++t) total += g.seg[t].K >> 5;

  int s = 0, kk = 0;
  stage(0, 0, 0);

  for (int step = 0; step < total; ++step) {
    const int buf = step & 1;
    int ns = s, nk = kk + 32;
    if (nk >= g.seg[ns].K) { ns++; nk = 0; }

    if (step + 1 < total) {
      stage(ns, nk, buf ^ 1);                 // 3 more asyncs in flight
#if USE_ASYNC_LDS
      __builtin_amdgcn_s_wait_asynccnt(3);    // retire current tile only
#endif
    }
#if USE_ASYNC_LDS
    else {
      __builtin_amdgcn_s_wait_asynccnt(0);
    }
#endif
    __syncthreads();                          // current tile visible to all waves

    const unsigned short* Ab = &As[buf * (64 * 40)];
    const unsigned short* Bb = &Bs[buf * (128 * 40)];

    // ---- fragments (ISA 7.12.2 16-bit layouts) ----
    v16bf aF[2], bF[2];
    #pragma unroll
    for (int i = 0; i < 2; ++i) {
      // A 16x32: lanes 0-15 row M, K = 0..7 & 16..23; lanes 16-31 K+8
      const unsigned short* p = &Ab[(wm * 32 + i * 16 + laneLo) * 40 + half * 8];
      Frag32 f; f.a = *(const uint4*)p; f.b = *(const uint4*)(p + 16);
      aF[i] = __builtin_bit_cast(v16bf, f);
    }
    #pragma unroll
    for (int j = 0; j < 2; ++j) {
      // B 32x16: lanes 0-15 col N, K = 0..7 & 16..23; lanes 16-31 K+8
      const unsigned short* q = &Bb[(wn * 32 + j * 16 + laneLo) * 40 + half * 8];
      Frag32 f; f.a = *(const uint4*)q; f.b = *(const uint4*)(q + 16);
      bF[j] = __builtin_bit_cast(v16bf, f);
    }

    #pragma unroll
    for (int i = 0; i < 2; ++i)
      #pragma unroll
      for (int j = 0; j < 2; ++j)
        acc[i][j] = __builtin_amdgcn_wmma_f32_16x16x32_bf16(
            false, aF[i], false, bF[j], (short)0, acc[i][j], false, false);

    __syncthreads();                          // all waves done with buf
    s = ns; kk = nk;
  }

  // ---- fused epilogue: u = c1*u_prev + c2*acc ; h = tanh(u) ----
  const float* __restrict__ up = g.uprev;
  float* __restrict__ oh = g.out + g.hcol;
  #pragma unroll
  for (int i = 0; i < 2; ++i) {
    #pragma unroll
    for (int j = 0; j < 2; ++j) {
      const int r0 = blockM + wm * 32 + i * 16 + half * 8;  // C: VGPR v -> M=v(+8)
      const int c  = blockN + wn * 32 + j * 16 + laneLo;    // C: N = lane (lo)
      float uv[8];
      #pragma unroll
      for (int v = 0; v < 8; ++v)                            // batched loads
        uv[v] = __builtin_nontemporal_load(&up[(size_t)(r0 + v) * g.N + c]);
      #pragma unroll
      for (int v = 0; v < 8; ++v) {
        const size_t ro = (size_t)(r0 + v) * OUT_STRIDE + c;
        float un = fmaf(g.c1, uv[v], g.c2 * acc[i][j][v]);
        __builtin_nontemporal_store(tanh_fast(un), &oh[ro]);           // h
        __builtin_nontemporal_store(un,            &oh[ro + U_SECT]);  // u
      }
    }
  }
}

// ---------------------------------------------------------------------------
extern "C" void kernel_launch(void* const* d_in, const int* in_sizes, int n_in,
                              void* d_out, int out_size, void* d_ws, size_t ws_size,
                              hipStream_t stream)
{
  (void)in_sizes; (void)n_in; (void)out_size; (void)ws_size;

  // Flat input order (setup_inputs dict insertion order, recursive):
  //  0-3: xl_{pose,enc,force,prox}   4-7: xr_*
  //  8-17: state_l {h_pose,u_pose,h_enc,u_enc,h_force,u_force,h_prox,u_prox,h_slow,u_slow}
  // 18-27: state_r (same)   28: state_u.h   29: state_u.u
  // 30-48: w_l {i2f,f2f,f2s,s2f}x{pose,enc,force,prox}, s2s(46), s2u(47), u2s(48)
  // 49-67: w_r (same)       68: u2u   69: kernel_att   70: bias_att
  const float* IN[71];
  for (int i = 0; i < 71; ++i) IN[i] = (const float*)d_in[i];
  float* out = (float*)d_out;

  char* ws = (char*)d_ws;
  size_t off = 0;
  auto walloc = [&](size_t bytes) -> void* {
    void* p = ws + off;
    off = (off + bytes + 255) & ~(size_t)255;
    return p;
  };

  float* attL = (float*)walloc((size_t)BROWS * 3 * sizeof(float));
  float* attR = (float*)walloc((size_t)BROWS * 3 * sizeof(float));

  const int xdim[4]   = {64, 256, 64, 128};
  const int xshift[4] = {6, 8, 6, 7};
  unsigned short *xb[2][4], *hb[2][4], *hsb[2], *sab[2][3], *hub;

  PackAArgs pa{}; int na = 0;
  for (int s = 0; s < 2; ++s) {
    const int xbase = s ? 4 : 0, sbase = s ? 18 : 8;
    float* attS = s ? attR : attL;
    for (int m = 0; m < 4; ++m) {
      xb[s][m] = (unsigned short*)walloc((size_t)BROWS * xdim[m] * 2);
      pa.j[na++] = { IN[xbase + m], xb[s][m], nullptr, 0, xshift[m] };
    }
    for (int m = 0; m < 4; ++m) {
      hb[s][m] = (unsigned short*)walloc((size_t)BROWS * 512 * 2);
      pa.j[na++] = { IN[sbase + 2 * m], hb[s][m], nullptr, 0, 9 };
    }
    hsb[s] = (unsigned short*)walloc((size_t)BROWS * 256 * 2);
    pa.j[na++] = { IN[sbase + 8], hsb[s], nullptr, 0, 8 };
    for (int t = 0; t < 3; ++t) {  // a0*h_enc, a1*h_force, a2*h_prox
      sab[s][t] = (unsigned short*)walloc((size_t)BROWS * 512 * 2);
      pa.j[na++] = { IN[sbase + 2 * (t + 1)], sab[s][t], attS, t, 9 };
    }
  }
  hub = (unsigned short*)walloc((size_t)BROWS * 256 * 2);
  pa.j[na++] = { IN[28], hub, nullptr, 0, 8 };   // na == 25

  PackWArgs pw{}; int nw = 0;
  auto addW = [&](const float* w, int kshift, int N) -> unsigned short* {
    unsigned short* wt = (unsigned short*)walloc(((size_t)N << kshift) * 2);
    pw.j[nw++] = { w, wt, kshift, N };
    return wt;
  };

  GemmArgs G[11]; int ng = 0;
  for (int s = 0; s < 2; ++s) {
    const int sbase = s ? 18 : 8, wbase = s ? 49 : 30;
    const int hbase = s ? 2304 : 0;
    for (int m = 0; m < 4; ++m) {           // fast: x@i2f + h@f2f + hs@s2f
      GemmArgs g{};
      g.seg[0] = { xb[s][m], addW(IN[wbase + 4 * m + 0], xshift[m], 512), xdim[m] };
      g.seg[1] = { hb[s][m], addW(IN[wbase + 4 * m + 1], 9, 512), 512 };
      g.seg[2] = { hsb[s],   addW(IN[wbase + 4 * m + 3], 8, 512), 256 };
      g.nseg = 3; g.N = 512;
      g.uprev = IN[sbase + 2 * m + 1]; g.c1 = 0.5f; g.c2 = 0.5f;
      g.out = out; g.hcol = hbase + m * 512;
      G[ng++] = g;
    }
    {                                        // slow
      GemmArgs g{};
      g.seg[0] = { hb[s][0],  addW(IN[wbase + 2],         9, 256), 512 }; // f2s_pose
      g.seg[1] = { sab[s][0], addW(IN[wbase + 4 * 1 + 2], 9, 256), 512 }; // f2s_enc
      g.seg[2] = { sab[s][1], addW(IN[wbase + 4 * 2 + 2], 9, 256), 512 }; // f2s_force
      g.seg[3] = { sab[s][2], addW(IN[wbase + 4 * 3 + 2], 9, 256), 512 }; // f2s_prox
      g.seg[4] = { hsb[s],    addW(IN[wbase + 16],        8, 256), 256 }; // s2s
      g.seg[5] = { hub,       addW(IN[wbase + 18],        8, 256), 256 }; // u2s
      g.nseg = 6; g.N = 256;
      g.uprev = IN[sbase + 9]; g.c1 = 11.f / 12.f; g.c2 = 1.f / 12.f;
      g.out = out; g.hcol = s ? 4352 : 2048;
      G[ng++] = g;
    }
  }
  {                                          // union
    GemmArgs g{};
    g.seg[0] = { hsb[0], addW(IN[30 + 17], 8, 256), 256 };  // s2u_l
    g.seg[1] = { hsb[1], addW(IN[49 + 17], 8, 256), 256 };  // s2u_r
    g.seg[2] = { hub,    addW(IN[68],      8, 256), 256 };  // u2u
    g.nseg = 3; g.N = 256;
    g.uprev = IN[29]; g.c1 = 24.f / 25.f; g.c2 = 1.f / 25.f;
    g.out = out; g.hcol = 4608;
    G[ng++] = g;                                             // ng == 11, nw == 39
  }

  att_kernel<<<dim3(BROWS / 256, 2), 256, 0, stream>>>(
      IN[16], IN[26], IN[69], IN[70], attL, attR);
  packA_kernel<<<dim3((BROWS * 512) / 256, 25), 256, 0, stream>>>(pa);
  packW_kernel<<<dim3((512 * 512) / 256, 39), 256, 0, stream>>>(pw);
  for (int i = 0; i < ng; ++i)
    gemm_kernel<<<dim3(BROWS / 64, G[i].N / 128), 256, 0, stream>>>(G[i]);
}
